// MultiHeadAttention_49323404427741
// MI455X (gfx1250) — matrix operand order repaired and checked by reading, not compiled, not run on hardware
//
#include <hip/hip_runtime.h>

typedef __attribute__((ext_vector_type(16))) __bf16 v16bf;
typedef __attribute__((ext_vector_type(8)))  float  v8f;
typedef int v4i_ __attribute__((vector_size(16)));   // matches async-LDS builtin param

#define D_MODEL 1024
#define N_HEADS 16
#define DK      64
#define BATCH   2
#define SEQ     2048
#define MROWS   (BATCH * SEQ)   // 4096

#define AS1 __attribute__((address_space(1)))
#define AS3 __attribute__((address_space(3)))

#if __has_builtin(__builtin_amdgcn_global_load_async_to_lds_b128)
#define ASYNC_LDS 1
#endif

static __device__ __forceinline__ void sched_fence() {
#if __has_builtin(__builtin_amdgcn_sched_barrier)
    __builtin_amdgcn_sched_barrier(0);   // no instruction may be moved across
#endif
}

static __device__ __forceinline__ unsigned short f2bf(float f) {
    unsigned int u = __builtin_bit_cast(unsigned int, f);
    unsigned int r = u + 0x7FFFu + ((u >> 16) & 1u);  // round-to-nearest-even
    return (unsigned short)(r >> 16);
}

union Frag {            // 16 bf16 elements = 32 bytes = two uint4
    uint4 q[2];
    v16bf v;
};

static __device__ __forceinline__ void wait_async_lds() {
#ifdef ASYNC_LDS
#if __has_builtin(__builtin_amdgcn_s_wait_asynccnt)
    __builtin_amdgcn_s_wait_asynccnt(0);
#else
    asm volatile("s_wait_asynccnt 0x0" ::: "memory");
#endif
#endif
}

// Cooperative copy of a 128x32 bf16 B tile (8 KB) into LDS: 256 threads x 32B.
static __device__ __forceinline__ void copy_btile(
    const unsigned short* __restrict__ W, int n0, int k0,
    unsigned short (*Bt)[32], int tid)
{
    const int row = tid >> 1;
    const int kh  = (tid & 1) * 16;
    const unsigned short* src = W + (size_t)(n0 + row) * D_MODEL + k0 + kh;
    unsigned short* dst = &Bt[row][kh];
#ifdef ASYNC_LDS
    __builtin_amdgcn_global_load_async_to_lds_b128(
        (AS1 v4i_*)(src),     (AS3 v4i_*)(dst),     0, 0);
    __builtin_amdgcn_global_load_async_to_lds_b128(
        (AS1 v4i_*)(src + 8), (AS3 v4i_*)(dst + 8), 0, 0);
#else
    *(uint4*)(dst)     = *(const uint4*)(src);
    *(uint4*)(dst + 8) = *(const uint4*)(src + 8);
#endif
}

// ---------------------------------------------------------------------------
// Kernel 0: fp32 -> bf16 conversion
// ---------------------------------------------------------------------------
__global__ void cvt_f32_bf16(const float* __restrict__ in,
                             unsigned short* __restrict__ out, int n) {
    int i = blockIdx.x * blockDim.x + threadIdx.x;
    if (i < n) out[i] = f2bf(in[i]);
}

// ---------------------------------------------------------------------------
// Kernel 1: fused QKV projection.  Y = X @ W^T + b
//   Block = 8 waves = 128(M) x 128(N) tile.  B tile staged in LDS (double
//   buffered, async-to-LDS).  Per wave / k-step of 32: 2 global b128 (A) +
//   16 ds_load_b128 (all issued before WMMA via sched_barrier), then 8
//   back-to-back wmma.  blockIdx.y selects Q / K / V.
//   Q,K written [B,H,T,64]; V written [B,H,64,T].
// ---------------------------------------------------------------------------
__global__ __launch_bounds__(256) void proj_qkv(
    const unsigned short* __restrict__ X,
    const unsigned short* __restrict__ Wq,
    const unsigned short* __restrict__ Wk,
    const unsigned short* __restrict__ Wv,
    const float* __restrict__ bq, const float* __restrict__ bk,
    const float* __restrict__ bv,
    unsigned short* __restrict__ Q, unsigned short* __restrict__ K,
    unsigned short* __restrict__ Vt)
{
    __shared__ __align__(16) unsigned short Bt[2][128][32];

    const int z = blockIdx.y;
    const unsigned short* W  = (z == 0) ? Wq : ((z == 1) ? Wk : Wv);
    const float*          pb = (z == 0) ? bq : ((z == 1) ? bk : bv);

    const int tid  = threadIdx.x;
    const int ln   = tid & 31;
    const int wvid = tid >> 5;
    const int mBase = (blockIdx.x >> 3) * 128 + wvid * 16;  // 32 m-blocks
    const int nBase = (blockIdx.x & 7) * 128;               // 8 n-blocks
    const int mI = ln & 15, hh = ln >> 4;

    v8f acc[8];
    #pragma unroll
    for (int j = 0; j < 8; ++j)
        #pragma unroll
        for (int r = 0; r < 8; ++r) acc[j][r] = 0.0f;

    const unsigned short* arow = X + (size_t)(mBase + mI) * D_MODEL;

    copy_btile(W, nBase, 0, Bt[0], tid);
    wait_async_lds();
    __syncthreads();

    int cur = 0;
    for (int k0 = 0; k0 < D_MODEL; k0 += 32) {
        const int nxt = cur ^ 1;
        if (k0 + 32 < D_MODEL) copy_btile(W, nBase, k0 + 32, Bt[nxt], tid);

        Frag a;
        a.q[0] = *(const uint4*)(arow + k0 + 16 * hh);
        a.q[1] = *(const uint4*)(arow + k0 + 16 * hh + 8);

        // Batch all B-fragment LDS reads (kept live in distinct registers).
        Frag bf[8];
        #pragma unroll
        for (int j = 0; j < 8; ++j) {
            const unsigned short* bl = &Bt[cur][16 * j + mI][16 * hh];
            bf[j].q[0] = *(const uint4*)(bl);
            bf[j].q[1] = *(const uint4*)(bl + 8);
        }
        sched_fence();   // all loads issued before any WMMA
        #pragma unroll
        for (int j = 0; j < 8; ++j)
            acc[j] = __builtin_amdgcn_wmma_f32_16x16x32_bf16(
                false, a.v, false, bf[j].v, (short)0, acc[j], false, false);

        wait_async_lds();
        __syncthreads();
        cur = nxt;
    }

    #pragma unroll
    for (int j = 0; j < 8; ++j) {
        const int n = nBase + 16 * j + mI;
        const float bval = pb[n];
        const int h = n >> 6, d = n & 63;
        #pragma unroll
        for (int r = 0; r < 8; ++r) {
            const int m  = mBase + 8 * hh + r;
            const int bb = m >> 11, t = m & (SEQ - 1);
            const unsigned short o = f2bf(acc[j][r] + bval);
            if (z == 0)
                Q[((size_t)(bb * N_HEADS + h) * SEQ + t) * DK + d] = o;
            else if (z == 1)
                K[((size_t)(bb * N_HEADS + h) * SEQ + t) * DK + d] = o;
            else
                Vt[((size_t)(bb * N_HEADS + h) * DK + d) * SEQ + t] = o;
        }
    }
}

// ---------------------------------------------------------------------------
// Kernel 2: causal flash attention.  One wave per (b,h, 16-row q-tile).
//   Q,K: [B,H,T,64] bf16;  Vt: [B,H,64,T] bf16;  Ctx out: [B*T, 1024] bf16.
//   Key blocks of 32.  All global loads (K + V fragments) issued at the top
//   of the iteration (sched_barrier keeps them ahead of the WMMAs) so V-load
//   latency hides under the softmax VALU work.
// ---------------------------------------------------------------------------
__global__ __launch_bounds__(256) void attn_fwd(
    const unsigned short* __restrict__ Q,
    const unsigned short* __restrict__ Km,
    const unsigned short* __restrict__ Vt,
    unsigned short* __restrict__ Ctx)
{
    __shared__ __align__(16) unsigned short Pst[8][16][32];  // per-wave 16x32 tile

    const int ln   = threadIdx.x & 31;
    const int wvid = threadIdx.x >> 5;
    const int wid  = blockIdx.x * 8 + wvid;   // 4096 waves
    const int qt   = wid & 127;
    const int bh   = wid >> 7;                // 0..31
    const int t0   = qt * 16;
    const int mI = ln & 15, hh = ln >> 4;

    const unsigned short* Qb = Q  + (size_t)bh * SEQ * DK;
    const unsigned short* Kb = Km + (size_t)bh * SEQ * DK;
    const unsigned short* Vb = Vt + (size_t)bh * DK * SEQ;

    // Q A-fragments (persistent): d_k = 64 -> two 16x32 fragments
    Frag aQ[2];
    const unsigned short* qrow = Qb + (size_t)(t0 + mI) * DK;
    #pragma unroll
    for (int c = 0; c < 2; ++c) {
        aQ[c].q[0] = *(const uint4*)(qrow + 32 * c + 16 * hh);
        aQ[c].q[1] = *(const uint4*)(qrow + 32 * c + 16 * hh + 8);
    }

    v8f accO[4];
    float mrow[8], lrow[8];
    #pragma unroll
    for (int g = 0; g < 4; ++g)
        #pragma unroll
        for (int r = 0; r < 8; ++r) accO[g][r] = 0.0f;
    #pragma unroll
    for (int r = 0; r < 8; ++r) { mrow[r] = -1.0e30f; lrow[r] = 0.0f; }

    const float sc = 0.1803368801f;  // log2(e) / sqrt(64) : softmax in exp2 domain

    for (int s0 = 0; s0 <= t0 + 15; s0 += 32) {
        // ---- issue ALL global loads for this iteration up front ----
        Frag bK[2][2];
        #pragma unroll
        for (int j = 0; j < 2; ++j) {
            const unsigned short* krow = Kb + (size_t)(s0 + 16 * j + mI) * DK;
            bK[j][0].q[0] = *(const uint4*)(krow + 16 * hh);
            bK[j][0].q[1] = *(const uint4*)(krow + 16 * hh + 8);
            bK[j][1].q[0] = *(const uint4*)(krow + 32 + 16 * hh);
            bK[j][1].q[1] = *(const uint4*)(krow + 32 + 16 * hh + 8);
        }
        Frag bV[4];
        #pragma unroll
        for (int g = 0; g < 4; ++g) {
            const unsigned short* vrow =
                Vb + (size_t)(16 * g + mI) * SEQ + s0 + 16 * hh;
            bV[g].q[0] = *(const uint4*)(vrow);
            bV[g].q[1] = *(const uint4*)(vrow + 8);
        }
        sched_fence();   // all 16 global loads issued before first WMMA

        // ---- S = Q K^T for a 16x32 score tile (two 16x16 column groups) ----
        v8f s[2];
        #pragma unroll
        for (int j = 0; j < 2; ++j) {
            v8f zc;
            #pragma unroll
            for (int r = 0; r < 8; ++r) zc[r] = 0.0f;
            zc   = __builtin_amdgcn_wmma_f32_16x16x32_bf16(
                       false, aQ[0].v, false, bK[j][0].v, (short)0, zc, false, false);
            s[j] = __builtin_amdgcn_wmma_f32_16x16x32_bf16(
                       false, aQ[1].v, false, bK[j][1].v, (short)0, zc, false, false);
        }

        // ---- scale + causal mask + row max over this 32-key block ----
        float tmax[8];
        #pragma unroll
        for (int r = 0; r < 8; ++r) {
            const int row = t0 + 8 * hh + r;
            #pragma unroll
            for (int j = 0; j < 2; ++j) {
                const int col = s0 + 16 * j + mI;
                float v = s[j][r] * sc;
                s[j][r] = (col > row) ? -1.0e30f : v;
            }
            tmax[r] = fmaxf(s[0][r], s[1][r]);
        }
        #pragma unroll
        for (int off = 1; off < 16; off <<= 1)
            #pragma unroll
            for (int r = 0; r < 8; ++r)
                tmax[r] = fmaxf(tmax[r], __shfl_xor(tmax[r], off, 32));

        // ---- online softmax update ----
        float alpha[8], rs[8];
        #pragma unroll
        for (int r = 0; r < 8; ++r) {
            const float mn = fmaxf(mrow[r], tmax[r]);
            alpha[r] = exp2f(mrow[r] - mn);
            mrow[r]  = mn;
            const float p0 = exp2f(s[0][r] - mn);
            const float p1 = exp2f(s[1][r] - mn);
            s[0][r] = p0; s[1][r] = p1;
            rs[r] = p0 + p1;
        }
        #pragma unroll
        for (int off = 1; off < 16; off <<= 1)
            #pragma unroll
            for (int r = 0; r < 8; ++r)
                rs[r] += __shfl_xor(rs[r], off, 32);
        #pragma unroll
        for (int r = 0; r < 8; ++r) lrow[r] = lrow[r] * alpha[r] + rs[r];
        #pragma unroll
        for (int g = 0; g < 4; ++g)
            #pragma unroll
            for (int r = 0; r < 8; ++r) accO[g][r] *= alpha[r];

        // ---- C-fragment P -> LDS -> A-fragment (layout shuffle) ----
        #pragma unroll
        for (int j = 0; j < 2; ++j)
            #pragma unroll
            for (int r = 0; r < 8; ++r)
                Pst[wvid][8 * hh + r][16 * j + mI] = f2bf(s[j][r]);
        asm volatile("s_wait_dscnt 0" ::: "memory");
        Frag aP;
        aP.q[0] = *(const uint4*)&Pst[wvid][mI][16 * hh];
        aP.q[1] = *(const uint4*)&Pst[wvid][mI][16 * hh + 8];

        // ---- O += P V with preloaded V fragments ----
        #pragma unroll
        for (int g = 0; g < 4; ++g)
            accO[g] = __builtin_amdgcn_wmma_f32_16x16x32_bf16(
                false, aP.v, false, bV[g].v, (short)0, accO[g], false, false);
    }

    // ---- normalize and write ctx (heads re-merged: [B*T, 1024]) ----
    float inv[8];
    #pragma unroll
    for (int r = 0; r < 8; ++r) inv[r] = 1.0f / lrow[r];
    const int bb = bh >> 4, h = bh & 15;
    #pragma unroll
    for (int g = 0; g < 4; ++g)
        #pragma unroll
        for (int r = 0; r < 8; ++r) {
            const int t = t0 + 8 * hh + r;
            Ctx[(size_t)(bb * SEQ + t) * D_MODEL + h * DK + 16 * g + mI] =
                f2bf(accO[g][r] * inv[r]);
        }
}

// ---------------------------------------------------------------------------
// Kernel 3: output projection.  Out = Ctx @ Wo^T + bo  (fp32 out)
//   Same LDS-staged block-tiled GEMM as proj_qkv.
// ---------------------------------------------------------------------------
__global__ __launch_bounds__(256) void out_proj(
    const unsigned short* __restrict__ Xc,
    const unsigned short* __restrict__ Wo,
    const float* __restrict__ bo,
    float* __restrict__ Out)
{
    __shared__ __align__(16) unsigned short Bt[2][128][32];

    const int tid  = threadIdx.x;
    const int ln   = tid & 31;
    const int wvid = tid >> 5;
    const int mBase = (blockIdx.x >> 3) * 128 + wvid * 16;
    const int nBase = (blockIdx.x & 7) * 128;
    const int mI = ln & 15, hh = ln >> 4;

    v8f acc[8];
    #pragma unroll
    for (int j = 0; j < 8; ++j)
        #pragma unroll
        for (int r = 0; r < 8; ++r) acc[j][r] = 0.0f;

    const unsigned short* arow = Xc + (size_t)(mBase + mI) * D_MODEL;

    copy_btile(Wo, nBase, 0, Bt[0], tid);
    wait_async_lds();
    __syncthreads();

    int cur = 0;
    for (int k0 = 0; k0 < D_MODEL; k0 += 32) {
        const int nxt = cur ^ 1;
        if (k0 + 32 < D_MODEL) copy_btile(Wo, nBase, k0 + 32, Bt[nxt], tid);

        Frag a;
        a.q[0] = *(const uint4*)(arow + k0 + 16 * hh);
        a.q[1] = *(const uint4*)(arow + k0 + 16 * hh + 8);

        Frag bf[8];
        #pragma unroll
        for (int j = 0; j < 8; ++j) {
            const unsigned short* bl = &Bt[cur][16 * j + mI][16 * hh];
            bf[j].q[0] = *(const uint4*)(bl);
            bf[j].q[1] = *(const uint4*)(bl + 8);
        }
        sched_fence();   // all loads issued before any WMMA
        #pragma unroll
        for (int j = 0; j < 8; ++j)
            acc[j] = __builtin_amdgcn_wmma_f32_16x16x32_bf16(
                false, a.v, false, bf[j].v, (short)0, acc[j], false, false);

        wait_async_lds();
        __syncthreads();
        cur = nxt;
    }

    #pragma unroll
    for (int j = 0; j < 8; ++j) {
        const int n = nBase + 16 * j + mI;
        const float bval = bo[n];
        #pragma unroll
        for (int r = 0; r < 8; ++r) {
            const int m = mBase + 8 * hh + r;
            Out[(size_t)m * D_MODEL + n] = acc[j][r] + bval;
        }
    }
}

// ---------------------------------------------------------------------------
extern "C" void kernel_launch(void* const* d_in, const int* in_sizes, int n_in,
                              void* d_out, int out_size, void* d_ws, size_t ws_size,
                              hipStream_t stream) {
    const float* x  = (const float*)d_in[0];
    const float* Wq = (const float*)d_in[1];
    const float* bq = (const float*)d_in[2];
    const float* Wk = (const float*)d_in[3];
    const float* bk = (const float*)d_in[4];
    const float* Wv = (const float*)d_in[5];
    const float* bv = (const float*)d_in[6];
    const float* Wo = (const float*)d_in[7];
    const float* bo = (const float*)d_in[8];

    // workspace carve-up (~50.5 MB total)
    char* p = (char*)d_ws;
    auto take = [&](size_t bytes) {
        char* r = p;
        p += (bytes + 255) & ~(size_t)255;
        return r;
    };
    unsigned short* xbf  = (unsigned short*)take((size_t)MROWS * D_MODEL * 2);
    unsigned short* wqbf = (unsigned short*)take((size_t)D_MODEL * D_MODEL * 2);
    unsigned short* wkbf = (unsigned short*)take((size_t)D_MODEL * D_MODEL * 2);
    unsigned short* wvbf = (unsigned short*)take((size_t)D_MODEL * D_MODEL * 2);
    unsigned short* wobf = (unsigned short*)take((size_t)D_MODEL * D_MODEL * 2);
    unsigned short* Qbf  = (unsigned short*)take((size_t)MROWS * D_MODEL * 2);
    unsigned short* Kbf  = (unsigned short*)take((size_t)MROWS * D_MODEL * 2);
    unsigned short* Vtbf = (unsigned short*)take((size_t)MROWS * D_MODEL * 2);
    unsigned short* Ctx  = (unsigned short*)take((size_t)MROWS * D_MODEL * 2);

    const int nx = MROWS * D_MODEL;     // 4M
    const int nw = D_MODEL * D_MODEL;   // 1M
    cvt_f32_bf16<<<(nx + 255) / 256, 256, 0, stream>>>(x,  xbf,  nx);
    cvt_f32_bf16<<<(nw + 255) / 256, 256, 0, stream>>>(Wq, wqbf, nw);
    cvt_f32_bf16<<<(nw + 255) / 256, 256, 0, stream>>>(Wk, wkbf, nw);
    cvt_f32_bf16<<<(nw + 255) / 256, 256, 0, stream>>>(Wv, wvbf, nw);
    cvt_f32_bf16<<<(nw + 255) / 256, 256, 0, stream>>>(Wo, wobf, nw);

    proj_qkv<<<dim3(256, 3, 1), 256, 0, stream>>>(
        xbf, wqbf, wkbf, wvbf, bq, bk, bv, Qbf, Kbf, Vtbf);

    attn_fwd<<<512, 256, 0, stream>>>(Qbf, Kbf, Vtbf, Ctx);

    out_proj<<<256, 256, 0, stream>>>(Ctx, wobf, bo, (float*)d_out);
}
